// OrthoLinear_833223655699
// MI455X (gfx1250) — matrix operand order repaired and checked by reading, not compiled, not run on hardware
//
#include <hip/hip_runtime.h>
#include <hip/hip_bf16.h>

#define OUT_F 2048
#define IN_F  2048

typedef __attribute__((ext_vector_type(16))) __bf16 v16bf;
typedef __attribute__((ext_vector_type(8)))  __bf16 v8bf;
typedef __attribute__((ext_vector_type(8)))  float  v8f;

// ---------------------------------------------------------------------------
// Float -> BF16, round-to-nearest-even (bit trick, no header dependency).
// ---------------------------------------------------------------------------
static __device__ __forceinline__ unsigned short f2bf(float f) {
    union { float f; unsigned u; } v; v.f = f;
    unsigned r = v.u + 0x7FFFu + ((v.u >> 16) & 1u);
    return (unsigned short)(r >> 16);
}

// ---------------------------------------------------------------------------
// Kernel 1: dequant int4-packed weights -> fp32 W in workspace.
// packed[o][j] holds two nibbles: low -> col 2j, high -> col 2j+1.
// ---------------------------------------------------------------------------
__global__ void k_dequant(const int* __restrict__ packed,
                          const float* __restrict__ scales,
                          float* __restrict__ Wf) {
    int idx = blockIdx.x * blockDim.x + threadIdx.x;     // 0 .. OUT_F*IN_F/2-1
    if (idx >= OUT_F * (IN_F / 2)) return;
    int o = idx >> 10;                                   // IN_F/2 == 1024
    int b = packed[idx];
    float s = scales[o];
    float2 w;
    w.x = (float)((b        & 0xF) - 8) * s;
    w.y = (float)(((b >> 4) & 0xF) - 8) * s;
    reinterpret_cast<float2*>(Wf)[idx] = w;              // floats o*2048 + 2j, 2j+1
}

// ---------------------------------------------------------------------------
// Kernel 2: COO residual scatter-add into fp32 W (global_atomic_add_f32).
// ---------------------------------------------------------------------------
__global__ void k_scatter(const float* __restrict__ vals,
                          const int* __restrict__ rows,
                          const int* __restrict__ cols,
                          float* __restrict__ Wf, int nnz) {
    int i = blockIdx.x * blockDim.x + threadIdx.x;
    if (i >= nnz) return;
    atomicAdd(&Wf[(size_t)rows[i] * IN_F + cols[i]], vals[i]);
}

// ---------------------------------------------------------------------------
// Kernel 3: fp32 -> bf16 conversion (2 elements/thread, packed dword stores).
// ---------------------------------------------------------------------------
__global__ void k_cvt_bf16(const float* __restrict__ src,
                           unsigned* __restrict__ dst, int n2) {
    int i = blockIdx.x * blockDim.x + threadIdx.x;
    if (i >= n2) return;
    float2 v = reinterpret_cast<const float2*>(src)[i];
    dst[i] = (unsigned)f2bf(v.x) | ((unsigned)f2bf(v.y) << 16);
}

// ---------------------------------------------------------------------------
// Kernel 4: bf16 WMMA GEMM.  out[t][o] = sum_k X[t][k] * W[o][k]
//   block = 128 threads = 4 waves (wave32), block tile 128(M) x 128(N)
//   wave tile 64x64 -> 4x4 accumulator tiles of 16x16 (v8f each)
//   K-step 32 -> one v_wmma_f32_16x16x32_bf16 per tile per step.
//   All 8 fragments (4xA, 4xB) are preloaded into distinct registers before
//   the 16-wmma block so the 16 b128 loads of a step (and, with unroll-2,
//   the next step) can stay in flight behind the matrix ops.
//
// Fragment layouts (ISA 7.12.2, wave32):
//   A 16x32 bf16: lane L (L<16: M=L, Kbase=0 / L>=16: M=L-16, Kbase=8):
//       elements 0..7  = K = Kbase .. Kbase+7
//       elements 8..15 = K = Kbase+16 .. Kbase+23
//   B 32x16 bf16: lane L (L<16: N=L, Kbase=0 / L>=16: N=L-16, Kbase=8):
//       same K order as A (Kbase..Kbase+7, Kbase+16..Kbase+23) along a W row.
//   C/D 16x16 f32: VGPR r, lanes 0-15: (M=r, N=lane); lanes 16-31: (M=r+8).
// ---------------------------------------------------------------------------
__global__ __launch_bounds__(128) void k_gemm(const __bf16* __restrict__ X,
                                              const __bf16* __restrict__ W,
                                              float* __restrict__ out) {
    const int lane  = threadIdx.x & 31;
    const int wid   = threadIdx.x >> 5;
    const int waveM = wid >> 1;
    const int waveN = wid & 1;
    const int rowBase = blockIdx.y * 128 + waveM * 64;   // token rows
    const int colBase = blockIdx.x * 128 + waveN * 64;   // output columns
    const int lhalf = lane >> 4;                         // 0 | 1
    const int l16   = lane & 15;

    // Per-lane base pointers (K advances by +32 elements per step).
    const __bf16* pa[4];
    const __bf16* pb[4];
#pragma unroll
    for (int ms = 0; ms < 4; ++ms)
        pa[ms] = X + (size_t)(rowBase + ms * 16 + l16) * IN_F + lhalf * 8;
#pragma unroll
    for (int ns = 0; ns < 4; ++ns)
        pb[ns] = W + (size_t)(colBase + ns * 16 + l16) * IN_F + lhalf * 8;

    v8f acc[4][4];
#pragma unroll
    for (int i = 0; i < 4; ++i)
#pragma unroll
        for (int j = 0; j < 4; ++j)
            acc[i][j] = (v8f){0.f, 0.f, 0.f, 0.f, 0.f, 0.f, 0.f, 0.f};

#pragma unroll 2
    for (int k = 0; k < IN_F; k += 32) {
        v16bf af[4], bf[4];
        // ---- preload all B fragments (two 8-elt K chunks per lane) ---------
#pragma unroll
        for (int ns = 0; ns < 4; ++ns) {
            const __bf16* p = pb[ns] + k;
            union { v16bf v; v8bf h[2]; } u;
            u.h[0] = *reinterpret_cast<const v8bf*>(p);
            u.h[1] = *reinterpret_cast<const v8bf*>(p + 16);
            bf[ns] = u.v;
        }
        // ---- preload all A fragments (two 8-elt K chunks per lane) ---------
#pragma unroll
        for (int ms = 0; ms < 4; ++ms) {
            const __bf16* p = pa[ms] + k;
            union { v16bf v; v8bf h[2]; } u;
            u.h[0] = *reinterpret_cast<const v8bf*>(p);        // K = kb..kb+7
            u.h[1] = *reinterpret_cast<const v8bf*>(p + 16);   // K = kb+16..kb+23
            af[ms] = u.v;
        }
        // ---- 16 matrix ops --------------------------------------------------
#pragma unroll
        for (int ms = 0; ms < 4; ++ms)
#pragma unroll
            for (int ns = 0; ns < 4; ++ns)
                acc[ms][ns] = __builtin_amdgcn_wmma_f32_16x16x32_bf16(
                    /*neg_a=*/false, af[ms], /*neg_b=*/false, bf[ns],
                    /*c_mod=*/(short)0, acc[ms][ns],
                    /*reuse_a=*/false, /*reuse_b=*/false);
    }

    // ---- Store C tiles ------------------------------------------------------
#pragma unroll
    for (int ms = 0; ms < 4; ++ms) {
        int r0 = rowBase + ms * 16 + lhalf * 8;
#pragma unroll
        for (int ns = 0; ns < 4; ++ns) {
            int c = colBase + ns * 16 + l16;
#pragma unroll
            for (int r = 0; r < 8; ++r)
                out[(size_t)(r0 + r) * OUT_F + c] = acc[ms][ns][r];
        }
    }
}

// ---------------------------------------------------------------------------
// Launch: dequant -> scatter -> convert(W) -> convert(x) -> WMMA GEMM
// Workspace layout:  [0,16MB) W fp32 | [16,24MB) W bf16 | [24,88MB) x bf16
// ---------------------------------------------------------------------------
extern "C" void kernel_launch(void* const* d_in, const int* in_sizes, int n_in,
                              void* d_out, int out_size, void* d_ws, size_t ws_size,
                              hipStream_t stream) {
    const float* x      = (const float*)d_in[0];
    const int*   packed = (const int*)  d_in[1];
    const float* scales = (const float*)d_in[2];
    const float* vals   = (const float*)d_in[3];
    const int*   rows   = (const int*)  d_in[4];
    const int*   cols   = (const int*)  d_in[5];
    float*       out    = (float*)d_out;

    const int nnz   = in_sizes[3];
    const int xelts = in_sizes[0];                 // 4*4096*2048 = 33554432
    const int ntok  = xelts / IN_F;                // 16384

    char* ws = (char*)d_ws;
    float*    Wf = (float*)ws;                                          // 16 MB
    unsigned* Wb = (unsigned*)(ws + (size_t)OUT_F * IN_F * 4);          //  8 MB
    unsigned* Xb = (unsigned*)(ws + (size_t)OUT_F * IN_F * 4
                                  + (size_t)OUT_F * IN_F * 2);          // 64 MB

    // 1) dequantize int4 base into fp32 W
    k_dequant<<<(OUT_F * (IN_F / 2) + 255) / 256, 256, 0, stream>>>(packed, scales, Wf);
    // 2) scatter sparse residual (fp32 atomics)
    k_scatter<<<(nnz + 255) / 256, 256, 0, stream>>>(vals, rows, cols, Wf, nnz);
    // 3) convert W and x to bf16
    k_cvt_bf16<<<(OUT_F * IN_F / 2 + 255) / 256, 256, 0, stream>>>(Wf, Wb, OUT_F * IN_F / 2);
    k_cvt_bf16<<<(xelts / 2 + 255) / 256, 256, 0, stream>>>(x, Xb, xelts / 2);
    // 4) WMMA GEMM: grid = (N/128, M/128), 128 threads/block
    k_gemm<<<dim3(OUT_F / 128, ntok / 128), 128, 0, stream>>>(
        (const __bf16*)Xb, (const __bf16*)Wb, out);
}
